// SelfAttention_12481174962287
// MI455X (gfx1250) — matrix operand order repaired and checked by reading, not compiled, not run on hardware
//
#include <hip/hip_runtime.h>
#include <hip/hip_bf16.h>

// ---------------------------------------------------------------------------
// Self-attention (B=2,S=2048,D=2048,H=16,HD=128) for gfx1250 (MI455X).
// bf16 WMMA everywhere, fp32 softmax/RoPE.
// Double-buffered LDS tiles fed by GLOBAL_LOAD_ASYNC_TO_LDS (ASYNCcnt) when
// the toolchain exposes it; register-staged fallback otherwise.
// ---------------------------------------------------------------------------

typedef __attribute__((ext_vector_type(16))) __bf16        v16bf;
typedef __attribute__((ext_vector_type(8)))  float         v8f;
typedef __attribute__((ext_vector_type(8)))  unsigned int  v8u;
typedef __attribute__((ext_vector_type(4)))  int           v4i;

#define D_MODEL 2048
#define SEQ     2048
#define NH      16
#define HDIM    128
#define BATCH   2
#define MTOK    (BATCH * SEQ)   // 4096 tokens

#if __has_builtin(__builtin_amdgcn_global_load_async_to_lds_b128)
#define ASYNC_LDS 1
#else
#define ASYNC_LDS 0
#endif

// ---------------- helpers ----------------

__device__ __forceinline__ unsigned short f2bf(float f) {
    unsigned int u = __float_as_uint(f);
    u += 0x7FFFu + ((u >> 16) & 1u);   // round-to-nearest-even
    return (unsigned short)(u >> 16);
}
__device__ __forceinline__ float bf2f(unsigned short h) {
    return __uint_as_float(((unsigned int)h) << 16);
}

__device__ __forceinline__ v8u pack2(int4 a, int4 b) {
    v8u r;
    r[0] = (unsigned)a.x; r[1] = (unsigned)a.y; r[2] = (unsigned)a.z; r[3] = (unsigned)a.w;
    r[4] = (unsigned)b.x; r[5] = (unsigned)b.y; r[6] = (unsigned)b.z; r[7] = (unsigned)b.w;
    return r;
}
// Build a 16-element bf16 WMMA fragment from two contiguous 8-element groups.
__device__ __forceinline__ v16bf frag16(const unsigned short* p0, const unsigned short* p1) {
    int4 a = *(const int4*)p0;
    int4 b = *(const int4*)p1;
    return __builtin_bit_cast(v16bf, pack2(a, b));
}
__device__ __forceinline__ v8f wmma_bf16(v16bf a, v16bf b, v8f c) {
    return __builtin_amdgcn_wmma_f32_16x16x32_bf16(
        /*neg_a=*/false, a, /*neg_b=*/false, b,
        /*c_mod=*/(short)0, c, /*reuse_a=*/false, /*reuse_b=*/false);
}

// 16-byte global -> LDS copy: async-to-LDS on gfx1250 toolchains that expose it.
__device__ __forceinline__ void cp16_g2l(const unsigned short* gsrc, unsigned short* ldst) {
#if ASYNC_LDS
    __builtin_amdgcn_global_load_async_to_lds_b128(
        (__attribute__((address_space(1))) v4i*)gsrc,
        (__attribute__((address_space(3))) v4i*)ldst,
        0, 0);
#else
    *(int4*)ldst = *(const int4*)gsrc;
#endif
}
__device__ __forceinline__ void async_wait0() {
#if ASYNC_LDS
#if __has_builtin(__builtin_amdgcn_s_wait_asynccnt)
    __builtin_amdgcn_s_wait_asynccnt(0);
#else
    asm volatile("s_wait_asynccnt 0x0" ::: "memory");
#endif
#endif
}
__device__ __forceinline__ void sched_fence() {
#if __has_builtin(__builtin_amdgcn_sched_barrier)
    __builtin_amdgcn_sched_barrier(0);
#endif
}

// ---------------- fp32 -> bf16 conversion ----------------

__global__ __launch_bounds__(256) void cvt_f32_bf16(const float* __restrict__ in,
                                                    unsigned short* __restrict__ out,
                                                    int n) {
    int i = (blockIdx.x * 256 + threadIdx.x) * 4;
    if (i + 3 < n) {
        float4 v = *(const float4*)(in + i);
        ushort4 o;
        o.x = f2bf(v.x); o.y = f2bf(v.y); o.z = f2bf(v.z); o.w = f2bf(v.w);
        *(ushort4*)(out + i) = o;
    }
}

// ---------------- GEMM: C[M,N] = A[M,K] @ W[N,K]^T + bias ----------------
// Double-buffered LDS (A + W tiles), async-to-LDS fills, one barrier/k-step.
// mode 0: write bf16 in [B,H,S,HD] layout (QKV).  mode 1: write fp32 row-major.

__global__ __launch_bounds__(256) void gemm_bf16_wmma(
    const unsigned short* __restrict__ A,    // [M, K] bf16
    const unsigned short* __restrict__ W,    // [N, K] bf16 (torch Linear weight)
    const float* __restrict__ bias,          // [N]
    unsigned short* __restrict__ outBf,      // mode 0
    float* __restrict__ outF,                // mode 1
    int mode) {
    __shared__ unsigned short Atile[2][128 * 32];   // 16 KB
    __shared__ unsigned short Wtile[2][128 * 32];   // 16 KB

    const int tid  = threadIdx.x;
    const int wave = tid >> 5;
    const int lane = tid & 31;
    const int lrow = lane & 15;
    const int half = lane >> 4;
    const int m0   = blockIdx.y * 128;
    const int n0   = blockIdx.x * 128;

    v8f acc[8];
#pragma unroll
    for (int t = 0; t < 8; ++t) acc[t] = (v8f){0.f,0.f,0.f,0.f,0.f,0.f,0.f,0.f};

    // cooperative loader mapping: each thread owns 16 bf16 of A and of W per step
    const int ldr = tid >> 1;
    const int ldc = (tid & 1) * 16;
    const unsigned short* Asrc = A + (size_t)(m0 + ldr) * D_MODEL + ldc;
    const unsigned short* Wsrc = W + (size_t)(n0 + ldr) * D_MODEL + ldc;
    const int lofs = ldr * 32 + ldc;

    // prologue: stage k-step 0 into buffer 0
    cp16_g2l(Asrc,     &Atile[0][lofs]);
    cp16_g2l(Asrc + 8, &Atile[0][lofs + 8]);
    cp16_g2l(Wsrc,     &Wtile[0][lofs]);
    cp16_g2l(Wsrc + 8, &Wtile[0][lofs + 8]);
    async_wait0();
    __syncthreads();

    const int NSTEP = D_MODEL / 32;   // 64
    for (int ks = 0; ks < NSTEP; ++ks) {
        const int cur = ks & 1, nxt = cur ^ 1;
        const int knext = (ks + 1) * 32;
        if (ks + 1 < NSTEP) {
            // fire next tile straight into the alternate LDS buffer (ASYNCcnt)
            cp16_g2l(Asrc + knext,     &Atile[nxt][lofs]);
            cp16_g2l(Asrc + knext + 8, &Atile[nxt][lofs + 8]);
            cp16_g2l(Wsrc + knext,     &Wtile[nxt][lofs]);
            cp16_g2l(Wsrc + knext + 8, &Wtile[nxt][lofs + 8]);
            if (ks + 2 < NSTEP) {
                __builtin_prefetch(Asrc + knext + 32, 0, 1);  // global_prefetch_b8
                __builtin_prefetch(Wsrc + knext + 32, 0, 1);
            }
            sched_fence();   // keep the fills above the math
        }

        // A fragment 16x32: lane row = wave*16+lrow, K groups {8h..+7, 16+8h..+7}
        const unsigned short* arow = &Atile[cur][(wave * 16 + lrow) * 32 + 8 * half];
        v16bf af = frag16(arow, arow + 16);
        v16bf wf[8];
#pragma unroll
        for (int t = 0; t < 8; ++t) {
            const unsigned short* wp = &Wtile[cur][(t * 16 + lrow) * 32 + 8 * half];
            wf[t] = frag16(wp, wp + 16);
        }
#pragma unroll
        for (int t = 0; t < 8; ++t) acc[t] = wmma_bf16(af, wf[t], acc[t]);

        if (ks + 1 < NSTEP) {
            async_wait0();      // next tile resident
            __syncthreads();
        }
    }

#pragma unroll
    for (int t = 0; t < 8; ++t) {
        const int col = n0 + t * 16 + lrow;
        const float bv = bias[col];
#pragma unroll
        for (int r = 0; r < 8; ++r) {
            const int row = m0 + wave * 16 + r + 8 * half;
            float v = acc[t][r] + bv;
            if (mode == 0) {
                int b = row >> 11, s = row & (SEQ - 1);
                int h = col >> 7,  hd = col & (HDIM - 1);
                outBf[((size_t)((b * NH + h) * SEQ + s)) * HDIM + hd] = f2bf(v);
            } else {
                outF[(size_t)row * D_MODEL + col] = v;
            }
        }
    }
}

// ---------------- RoPE (with reference's intentional sin/cos swap) ----------------

__global__ __launch_bounds__(256) void rope_kernel(unsigned short* __restrict__ Q,
                                                   unsigned short* __restrict__ Kv) {
    int idx = blockIdx.x * 256 + threadIdx.x;          // over B*H*S*64
    int j  = idx & 63;
    int s  = (idx >> 6) & (SEQ - 1);
    int bh = idx >> 17;
    size_t base = ((size_t)bh * SEQ + s) * HDIM;
    // inv_freq = 10000^{-2j/128}
    float inv_freq = __expf(-(float)(2 * j) * (1.0f / HDIM) * 9.210340371976184f);
    float ang  = (float)s * inv_freq;
    float mulq = __sinf(ang);   // cos_t after the reference's swap
    float mulr = __cosf(ang);   // sin_t after the reference's swap
    {
        float a = bf2f(Q[base + j]);
        float b = bf2f(Q[base + j + 64]);
        Q[base + j]      = f2bf(a * mulq - b * mulr);
        Q[base + j + 64] = f2bf(b * mulq + a * mulr);
    }
    {
        float a = bf2f(Kv[base + j]);
        float b = bf2f(Kv[base + j + 64]);
        Kv[base + j]      = f2bf(a * mulq - b * mulr);
        Kv[base + j + 64] = f2bf(b * mulq + a * mulr);
    }
}

// ---------------- Flash attention: per (b,h), 128-row q tile, 32-key steps ----------------
// K tile: async-to-LDS double buffer. V tile: register transpose (pinned early).

__global__ __launch_bounds__(256) void flash_attn(
    const unsigned short* __restrict__ Q,
    const unsigned short* __restrict__ K,
    const unsigned short* __restrict__ V,
    unsigned short* __restrict__ Ctx) {            // token-major [B*S, D] bf16
    __shared__ unsigned short Kt[2][32 * 128];     // K tile row-major   (16 KB)
    __shared__ unsigned short Vt[2][128 * 32];     // V tile transposed  (16 KB)
    __shared__ unsigned short Pt[8 * 512];         // per-wave P scratch ( 8 KB)

    const int tid  = threadIdx.x;
    const int wave = tid >> 5;
    const int lane = tid & 31;
    const int lrow = lane & 15;
    const int half = lane >> 4;
    const int bh   = blockIdx.y;
    const int q0   = blockIdx.x * 128;

    const size_t head = (size_t)bh * SEQ * HDIM;
    const unsigned short* Qh = Q + head;
    const unsigned short* Kh = K + head;
    const unsigned short* Vh = V + head;

    // cooperative loader mapping: thread covers 2 segments of (row, 8 cols)
    const int lq0 = tid * 2, lq1 = tid * 2 + 1;
    const int lr0 = lq0 >> 4, lc0 = (lq0 & 15) * 8;
    const int lr1 = lq1 >> 4, lc1 = (lq1 & 15) * 8;

    // Q fragments: 16 rows x 128 k -> 4 fragments of 16x32
    v16bf qf[4];
    {
        const unsigned short* qrow = Qh + (size_t)(q0 + wave * 16 + lrow) * HDIM;
#pragma unroll
        for (int c = 0; c < 4; ++c) {
            const unsigned short* p = qrow + c * 32 + 8 * half;
            qf[c] = frag16(p, p + 16);
        }
    }

    v8f o[8];
#pragma unroll
    for (int t = 0; t < 8; ++t) o[t] = (v8f){0.f,0.f,0.f,0.f,0.f,0.f,0.f,0.f};
    float mrow[8], lsum[8];
#pragma unroll
    for (int r = 0; r < 8; ++r) { mrow[r] = -1e30f; lsum[r] = 0.0f; }

    const float scale = 0.08838834764831845f;   // 1/sqrt(128)

    // prologue: stage kv tile 0 into buffer 0 (K async, V transposed via regs)
    {
        cp16_g2l(Kh + (size_t)lr0 * HDIM + lc0, &Kt[0][lr0 * 128 + lc0]);
        cp16_g2l(Kh + (size_t)lr1 * HDIM + lc1, &Kt[0][lr1 * 128 + lc1]);
        union { int4 v; unsigned short s[8]; } u0, u1;
        u0.v = *(const int4*)(Vh + (size_t)lr0 * HDIM + lc0);
        u1.v = *(const int4*)(Vh + (size_t)lr1 * HDIM + lc1);
#pragma unroll
        for (int j = 0; j < 8; ++j) {
            Vt[0][(lc0 + j) * 32 + lr0] = u0.s[j];
            Vt[0][(lc1 + j) * 32 + lr1] = u1.s[j];
        }
        async_wait0();
    }
    __syncthreads();

    const int NSTEP = SEQ / 32;   // 64
    for (int ks = 0; ks < NSTEP; ++ks) {
        const int cur = ks & 1, nxt = cur ^ 1;
        const int kvn = (ks + 1) * 32;

        int4 rv0, rv1;
        if (ks + 1 < NSTEP) {
            // next K tile: async straight to LDS; next V tile: early global loads
            cp16_g2l(Kh + (size_t)(kvn + lr0) * HDIM + lc0, &Kt[nxt][lr0 * 128 + lc0]);
            cp16_g2l(Kh + (size_t)(kvn + lr1) * HDIM + lc1, &Kt[nxt][lr1 * 128 + lc1]);
            rv0 = *(const int4*)(Vh + (size_t)(kvn + lr0) * HDIM + lc0);
            rv1 = *(const int4*)(Vh + (size_t)(kvn + lr1) * HDIM + lc1);
            if (ks + 2 < NSTEP) {
                __builtin_prefetch(Kh + (size_t)(kvn + 32 + lr0) * HDIM + lc0, 0, 1);
                __builtin_prefetch(Vh + (size_t)(kvn + 32 + lr0) * HDIM + lc0, 0, 1);
            }
            sched_fence();   // pin the fills/loads above the math
        }

        // scores: Q(16x128) @ K_tile^T -> two 16x16 tiles (keys 0..15, 16..31)
        v8f s0 = (v8f){0.f,0.f,0.f,0.f,0.f,0.f,0.f,0.f};
        v8f s1 = (v8f){0.f,0.f,0.f,0.f,0.f,0.f,0.f,0.f};
#pragma unroll
        for (int c = 0; c < 4; ++c) {
            const unsigned short* kp0 = &Kt[cur][(0  + lrow) * 128 + c * 32 + 8 * half];
            const unsigned short* kp1 = &Kt[cur][(16 + lrow) * 128 + c * 32 + 8 * half];
            s0 = wmma_bf16(qf[c], frag16(kp0, kp0 + 16), s0);
            s1 = wmma_bf16(qf[c], frag16(kp1, kp1 + 16), s1);
        }

        // online softmax: rows live as (VGPR r, lane-half) per C-matrix layout
        float tm[8];
#pragma unroll
        for (int r = 0; r < 8; ++r) {
            s0[r] *= scale; s1[r] *= scale;
            tm[r] = fmaxf(s0[r], s1[r]);
        }
#pragma unroll
        for (int off = 1; off < 16; off <<= 1)
#pragma unroll
            for (int r = 0; r < 8; ++r)
                tm[r] = fmaxf(tm[r], __shfl_xor(tm[r], off, 32));

        float corr[8], rs[8];
#pragma unroll
        for (int r = 0; r < 8; ++r) {
            float mn = fmaxf(mrow[r], tm[r]);
            corr[r]  = __expf(mrow[r] - mn);
            mrow[r]  = mn;
            float p0 = __expf(s0[r] - mn);
            float p1 = __expf(s1[r] - mn);
            s0[r] = p0; s1[r] = p1;
            rs[r] = p0 + p1;
        }
#pragma unroll
        for (int off = 1; off < 16; off <<= 1)
#pragma unroll
            for (int r = 0; r < 8; ++r)
                rs[r] += __shfl_xor(rs[r], off, 32);
#pragma unroll
        for (int r = 0; r < 8; ++r) {
            lsum[r] = lsum[r] * corr[r] + rs[r];
#pragma unroll
            for (int t = 0; t < 8; ++t) o[t][r] *= corr[r];
        }

        // C-layout -> A-layout transpose of P through per-wave LDS scratch
        unsigned short* pw = &Pt[wave * 512];
#pragma unroll
        for (int r = 0; r < 8; ++r) {
            int row = r + 8 * half;
            pw[row * 32 + lrow]      = f2bf(s0[r]);
            pw[row * 32 + 16 + lrow] = f2bf(s1[r]);
        }
        __builtin_amdgcn_wave_barrier();   // DS ops in-order within the wave
        const unsigned short* pr = &pw[lrow * 32 + 8 * half];
        v16bf pf = frag16(pr, pr + 16);

        // PV: prefetch all 8 V fragments, then the WMMA burst
        v16bf vf[8];
#pragma unroll
        for (int t = 0; t < 8; ++t) {
            const unsigned short* vp = &Vt[cur][(t * 16 + lrow) * 32 + 8 * half];
            vf[t] = frag16(vp, vp + 16);
        }
#pragma unroll
        for (int t = 0; t < 8; ++t) o[t] = wmma_bf16(pf, vf[t], o[t]);

        // commit next V tile (transposed) to the alternate buffer
        if (ks + 1 < NSTEP) {
            union { int4 v; unsigned short s[8]; } u0, u1;
            u0.v = rv0; u1.v = rv1;
#pragma unroll
            for (int j = 0; j < 8; ++j) {
                Vt[nxt][(lc0 + j) * 32 + lr0] = u0.s[j];
                Vt[nxt][(lc1 + j) * 32 + lr1] = u1.s[j];
            }
            async_wait0();     // K tile resident
            __syncthreads();
        }
    }

    // normalize and store ctx token-major for the output projection
    const int b = bh >> 4, h = bh & 15;
#pragma unroll
    for (int r = 0; r < 8; ++r) {
        float rcp = 1.0f / lsum[r];
        int srow = q0 + wave * 16 + r + 8 * half;
        size_t token = (size_t)b * SEQ + srow;
#pragma unroll
        for (int t = 0; t < 8; ++t) {
            int col = h * HDIM + t * 16 + lrow;
            Ctx[token * D_MODEL + col] = f2bf(o[t][r] * rcp);
        }
    }
}

// ---------------- launch ----------------

extern "C" void kernel_launch(void* const* d_in, const int* in_sizes, int n_in,
                              void* d_out, int out_size, void* d_ws, size_t ws_size,
                              hipStream_t stream) {
    const float* X  = (const float*)d_in[0];
    const float* Wq = (const float*)d_in[1];
    const float* bq = (const float*)d_in[2];
    const float* Wk = (const float*)d_in[3];
    const float* bk = (const float*)d_in[4];
    const float* Wv = (const float*)d_in[5];
    const float* bv = (const float*)d_in[6];
    const float* Wo = (const float*)d_in[7];
    const float* bo = (const float*)d_in[8];
    float* out = (float*)d_out;

    unsigned short* ws   = (unsigned short*)d_ws;
    unsigned short* Xbf  = ws;                     // 4096*2048
    unsigned short* Wqbf = Xbf  + 8388608;         // 2048*2048 each
    unsigned short* Wkbf = Wqbf + 4194304;
    unsigned short* Wvbf = Wkbf + 4194304;
    unsigned short* Wobf = Wvbf + 4194304;
    unsigned short* Qbf  = Wobf + 4194304;         // [B,H,S,HD]
    unsigned short* Kbf  = Qbf  + 8388608;
    unsigned short* Vbf  = Kbf  + 8388608;
    unsigned short* Ctx  = Vbf  + 8388608;         // [B*S, D]

    // fp32 -> bf16 staging
    cvt_f32_bf16<<<8388608 / 1024, 256, 0, stream>>>(X,  Xbf,  8388608);
    cvt_f32_bf16<<<4194304 / 1024, 256, 0, stream>>>(Wq, Wqbf, 4194304);
    cvt_f32_bf16<<<4194304 / 1024, 256, 0, stream>>>(Wk, Wkbf, 4194304);
    cvt_f32_bf16<<<4194304 / 1024, 256, 0, stream>>>(Wv, Wvbf, 4194304);
    cvt_f32_bf16<<<4194304 / 1024, 256, 0, stream>>>(Wo, Wobf, 4194304);

    dim3 gg(D_MODEL / 128, MTOK / 128);   // (16, 32)
    gemm_bf16_wmma<<<gg, 256, 0, stream>>>(Xbf, Wqbf, bq, Qbf, nullptr, 0);
    gemm_bf16_wmma<<<gg, 256, 0, stream>>>(Xbf, Wkbf, bk, Kbf, nullptr, 0);
    gemm_bf16_wmma<<<gg, 256, 0, stream>>>(Xbf, Wvbf, bv, Vbf, nullptr, 0);

    rope_kernel<<<(BATCH * NH * SEQ * 64) / 256, 256, 0, stream>>>(Qbf, Kbf);

    flash_attn<<<dim3(SEQ / 128, BATCH * NH), 256, 0, stream>>>(Qbf, Kbf, Vbf, Ctx);

    gemm_bf16_wmma<<<gg, 256, 0, stream>>>(Ctx, Wobf, bo, nullptr, out, 1);
}
